// SimpleQueryGenerator_77034533421760
// MI455X (gfx1250) — matrix operand order repaired and checked by reading, not compiled, not run on hardware
//
#include <hip/hip_runtime.h>

// ---------------------------------------------------------------------------
// Types for CDNA5 WMMA (gfx1250, wave32)
// ---------------------------------------------------------------------------
typedef __bf16 bf16_t;
typedef __attribute__((ext_vector_type(16))) __bf16 v16bf;
typedef __attribute__((ext_vector_type(8)))  float  v8f;
typedef __attribute__((ext_vector_type(4)))  float  f32x4;   // 16-byte chunk

union Frag32B { v16bf v; f32x4 q[2]; };                      // 32 bytes = 16 bf16

#define LDS_STRIDE 40   // 32 bf16 payload + 8 bf16 pad per row (80B, 16B aligned)

// ---------------------------------------------------------------------------
// bf16 WMMA GEMM:  C[M,N] = act( sum_s A_s[M,K] * W_s[N,K]^T + bias0 + bias1 )
//   - block: 256 threads = 8 waves; C tile 128x128; wave tile 32x64 (2x4 WMMA)
//   - A tile staged through LDS; B fragments straight from (L2-resident) weights
//   - mode 0: relu -> bf16 out        (encoder layers)
//   - mode 1: fp32 out                 (LSTM gate pre-activations)
//   - mode 2: fp32 strided out + bf16  (projection -> d_out[b,t,:] + feedback)
// ---------------------------------------------------------------------------
__global__ __launch_bounds__(256) void wmma_gemm_kernel(
    const bf16_t* __restrict__ A0, const bf16_t* __restrict__ W0,
    const bf16_t* __restrict__ A1, const bf16_t* __restrict__ W1w,
    const float*  __restrict__ bias0, const float* __restrict__ bias1,
    int N, int K, int nA, int mode,
    float* __restrict__ outF, bf16_t* __restrict__ outB,
    int outFStride, int outFColOff)
{
    __shared__ bf16_t ldsA[128 * LDS_STRIDE];

    const int  wave  = threadIdx.x >> 5;
    const int  lane  = threadIdx.x & 31;
    const int  l0    = lane & 15;
    const bool hilane = lane >= 16;
    const int  waveM = wave & 3;     // 4 waves along M (32 rows each)
    const int  waveN = wave >> 2;    // 2 waves along N (64 cols each)
    const int  m0    = blockIdx.y * 128;
    const int  n0    = blockIdx.x * 128;

    v8f acc[2][4];
    #pragma unroll
    for (int mi = 0; mi < 2; ++mi)
        #pragma unroll
        for (int ni = 0; ni < 4; ++ni)
            acc[mi][ni] = v8f{};

    for (int s = 0; s < nA; ++s) {
        const bf16_t* __restrict__ A = (s == 0) ? A0 : A1;
        const bf16_t* __restrict__ W = (s == 0) ? W0 : W1w;

        for (int k0 = 0; k0 < K; k0 += 32) {
            __syncthreads();     // all waves done reading previous tile
            // cooperative A-tile load: 128 rows x 32 bf16 = 512 chunks of 16B
            #pragma unroll
            for (int c = 0; c < 2; ++c) {
                const int chunk = (int)threadIdx.x + c * 256;
                const int row = chunk >> 2;
                const int seg = chunk & 3;
                const f32x4* src =
                    (const f32x4*)(A + (size_t)(m0 + row) * K + k0 + seg * 8);
                *(f32x4*)&ldsA[row * LDS_STRIDE + seg * 8] = *src;
            }
            __syncthreads();

            // B fragments: lane n = n0+waveN*64+ni*16+l0 reads 16 consecutive
            // K-values of W row n (lanes 16-31 take K+16..K+31).
            Frag32B bfrag[4];
            #pragma unroll
            for (int ni = 0; ni < 4; ++ni) {
                const int n = n0 + waveN * 64 + ni * 16 + l0;
                const bf16_t* bp = W + (size_t)n * K + k0 + (hilane ? 16 : 0);
                bfrag[ni].q[0] = *(const f32x4*)bp;
                bfrag[ni].q[1] = *(const f32x4*)(bp + 8);
                if (k0 + 32 < K)
                    __builtin_prefetch((const void*)(bp + 32), 0, 1);
            }

            // A fragments from LDS (contiguous K, matching B):
            //   lanes 0-15: K 0-15
            //   lanes 16-31: K 16-31
            Frag32B afrag[2];
            #pragma unroll
            for (int mi = 0; mi < 2; ++mi) {
                const bf16_t* ap =
                    &ldsA[(waveM * 32 + mi * 16 + l0) * LDS_STRIDE + (hilane ? 16 : 0)];
                afrag[mi].q[0] = *(const f32x4*)ap;
                afrag[mi].q[1] = *(const f32x4*)(ap + 8);
            }

            #pragma unroll
            for (int mi = 0; mi < 2; ++mi)
                #pragma unroll
                for (int ni = 0; ni < 4; ++ni)
                    acc[mi][ni] = __builtin_amdgcn_wmma_f32_16x16x32_bf16(
                        false, afrag[mi].v, false, bfrag[ni].v,
                        (short)0, acc[mi][ni], false, false);
        }
    }

    // Epilogue. C layout: VGPR r -> M = r + 8*(lane>=16), N = lane&15.
    #pragma unroll
    for (int mi = 0; mi < 2; ++mi) {
        #pragma unroll
        for (int ni = 0; ni < 4; ++ni) {
            const int ncol = n0 + waveN * 64 + ni * 16 + l0;
            float bv = bias0 ? bias0[ncol] : 0.0f;
            if (bias1) bv += bias1[ncol];
            #pragma unroll
            for (int r = 0; r < 8; ++r) {
                const int mrow = m0 + waveM * 32 + mi * 16 + (hilane ? 8 : 0) + r;
                float v = acc[mi][ni][r] + bv;
                if (mode == 0) {
                    v = v > 0.0f ? v : 0.0f;
                    outB[(size_t)mrow * N + ncol] = (bf16_t)v;
                } else if (mode == 1) {
                    outF[(size_t)mrow * outFStride + outFColOff + ncol] = v;
                } else {
                    outF[(size_t)mrow * outFStride + outFColOff + ncol] = v;
                    outB[(size_t)mrow * N + ncol] = (bf16_t)v;
                }
            }
        }
    }
}

// ---------------------------------------------------------------------------
// LSTM cell elementwise update (gate order i, f, g, o as in PyTorch)
// ---------------------------------------------------------------------------
__global__ void lstm_cell_kernel(const float* __restrict__ g,
                                 float* __restrict__ c,
                                 bf16_t* __restrict__ hout,
                                 int Bn, int Hn)
{
    const int idx = blockIdx.x * blockDim.x + threadIdx.x;
    if (idx >= Bn * Hn) return;
    const int b = idx / Hn;
    const int h = idx - b * Hn;
    const float* gr = g + (size_t)b * 4 * Hn;
    float gi = gr[h];
    float gf = gr[Hn + h];
    float gg = gr[2 * Hn + h];
    float go = gr[3 * Hn + h];
    gi = 1.0f / (1.0f + __expf(-gi));
    gf = 1.0f / (1.0f + __expf(-gf));
    go = 1.0f / (1.0f + __expf(-go));
    const float cv = gf * c[idx] + gi * tanhf(gg);
    c[idx] = cv;
    hout[idx] = (bf16_t)(go * tanhf(cv));
}

// ---------------------------------------------------------------------------
// fp32 -> bf16 conversion (weights & input, runs inside the captured graph so
// every replay is deterministic)
// ---------------------------------------------------------------------------
__global__ void cvt_f32_bf16_kernel(const float* __restrict__ s,
                                    bf16_t* __restrict__ d, int n)
{
    const int i = blockIdx.x * blockDim.x + threadIdx.x;
    if (i < n) d[i] = (bf16_t)s[i];
}

__global__ void zero_words_kernel(unsigned int* __restrict__ p, int nWords)
{
    const int i = blockIdx.x * blockDim.x + threadIdx.x;
    if (i < nWords) p[i] = 0u;
}

// ---------------------------------------------------------------------------
// Host-side launch sequence
// ---------------------------------------------------------------------------
extern "C" void kernel_launch(void* const* d_in, const int* in_sizes, int n_in,
                              void* d_out, int out_size, void* d_ws, size_t ws_size,
                              hipStream_t stream)
{
    (void)in_sizes; (void)n_in; (void)out_size; (void)ws_size;
    constexpr int Bn = 2048, INn = 768, Hn = 1024, Vn = 1024, T = 50, G4 = 4096;

    const float* x_in = (const float*)d_in[0];
    const float* W1   = (const float*)d_in[1];
    const float* b1   = (const float*)d_in[2];
    const float* W2   = (const float*)d_in[3];
    const float* b2   = (const float*)d_in[4];
    const float* Wih0 = (const float*)d_in[5];
    const float* Whh0 = (const float*)d_in[6];
    const float* bih0 = (const float*)d_in[7];
    const float* bhh0 = (const float*)d_in[8];
    const float* Wih1 = (const float*)d_in[9];
    const float* Whh1 = (const float*)d_in[10];
    const float* bih1 = (const float*)d_in[11];
    const float* bhh1 = (const float*)d_in[12];
    const float* Wo   = (const float*)d_in[13];
    const float* bo   = (const float*)d_in[14];
    float* out = (float*)d_out;

    // ---- workspace carve-up (all sizes already multiples of 256B) ----------
    char*  ws  = (char*)d_ws;
    size_t off = 0;
    auto take = [&](size_t bytes) -> char* {
        char* p = ws + off;
        off = (off + bytes + 255) & ~(size_t)255;
        return p;
    };
    bf16_t* W1b   = (bf16_t*)take((size_t)Hn * INn * 2);
    bf16_t* W2b   = (bf16_t*)take((size_t)Hn * Hn  * 2);
    bf16_t* Wih0b = (bf16_t*)take((size_t)G4 * Hn  * 2);
    bf16_t* Whh0b = (bf16_t*)take((size_t)G4 * Hn  * 2);
    bf16_t* Wih1b = (bf16_t*)take((size_t)G4 * Hn  * 2);
    bf16_t* Whh1b = (bf16_t*)take((size_t)G4 * Hn  * 2);
    bf16_t* Wob   = (bf16_t*)take((size_t)Vn * Hn  * 2);
    bf16_t* x0b   = (bf16_t*)take((size_t)Bn * INn * 2);
    bf16_t* enc1b = (bf16_t*)take((size_t)Bn * Hn  * 2);
    bf16_t* xb    = (bf16_t*)take((size_t)Bn * Hn  * 2);   // step input (feedback)
    bf16_t* h0b   = (bf16_t*)take((size_t)Bn * Hn  * 2);   // state block start
    bf16_t* h1b   = (bf16_t*)take((size_t)Bn * Hn  * 2);
    float*  c0    = (float*) take((size_t)Bn * Hn  * 4);
    float*  c1    = (float*) take((size_t)Bn * Hn  * 4);
    float*  gbuf  = (float*) take((size_t)Bn * G4  * 4);   // shared gate buffer

    auto cvt = [&](const float* s, bf16_t* d, int n) {
        cvt_f32_bf16_kernel<<<(n + 255) / 256, 256, 0, stream>>>(s, d, n);
    };
    auto gemm = [&](const bf16_t* A0, const bf16_t* Wm0,
                    const bf16_t* A1, const bf16_t* Wm1,
                    const float* bs0, const float* bs1,
                    int N, int K, int nA, int mode,
                    float* oF, bf16_t* oB, int oStride, int oColOff) {
        dim3 grid(N / 128, Bn / 128);
        wmma_gemm_kernel<<<grid, 256, 0, stream>>>(
            A0, Wm0, A1, Wm1, bs0, bs1, N, K, nA, mode, oF, oB, oStride, oColOff);
    };

    // ---- one-time (per replay) conversions + state init --------------------
    cvt(W1,   W1b,   Hn * INn);
    cvt(W2,   W2b,   Hn * Hn);
    cvt(Wih0, Wih0b, G4 * Hn);
    cvt(Whh0, Whh0b, G4 * Hn);
    cvt(Wih1, Wih1b, G4 * Hn);
    cvt(Whh1, Whh1b, G4 * Hn);
    cvt(Wo,   Wob,   Vn * Hn);
    cvt(x_in, x0b,   Bn * INn);

    {   // zero h0b|h1b|c0|c1 (contiguous): Bn*Hn*(2+2+4+4) bytes
        const int nWords = Bn * Hn * 3;                    // /4 bytes per word
        zero_words_kernel<<<(nWords + 255) / 256, 256, 0, stream>>>(
            (unsigned int*)h0b, nWords);
    }

    // ---- encoder MLP --------------------------------------------------------
    gemm(x0b,   W1b, nullptr, nullptr, b1, nullptr, Hn, INn, 1, 0,
         nullptr, enc1b, 0, 0);
    gemm(enc1b, W2b, nullptr, nullptr, b2, nullptr, Hn, Hn,  1, 0,
         nullptr, xb, 0, 0);

    // ---- 50 recurrent steps (serialized via stream order) -------------------
    const int cellBlocks = (Bn * Hn + 255) / 256;
    for (int t = 0; t < T; ++t) {
        // LSTM layer 0 gates: xb*Wih0^T + h0*Whh0^T + (bih0+bhh0)
        gemm(xb, Wih0b, h0b, Whh0b, bih0, bhh0, G4, Hn, 2, 1,
             gbuf, nullptr, G4, 0);
        lstm_cell_kernel<<<cellBlocks, 256, 0, stream>>>(gbuf, c0, h0b, Bn, Hn);

        // LSTM layer 1 gates: h0*Wih1^T + h1*Whh1^T + (bih1+bhh1)
        gemm(h0b, Wih1b, h1b, Whh1b, bih1, bhh1, G4, Hn, 2, 1,
             gbuf, nullptr, G4, 0);
        lstm_cell_kernel<<<cellBlocks, 256, 0, stream>>>(gbuf, c1, h1b, Bn, Hn);

        // projection: out[b,t,:] = h1*Wo^T + bo ; bf16 copy feeds next step
        gemm(h1b, Wob, nullptr, nullptr, bo, nullptr, Vn, Hn, 1, 2,
             out, xb, T * Vn, t * Vn);
    }
}
